// DualAttention_57939108823207
// MI455X (gfx1250) — matrix operand order repaired and checked by reading, not compiled, not run on hardware
//
#include <hip/hip_runtime.h>
#include <hip/hip_bf16.h>
#include <math.h>

// ---------------------------------------------------------------------------
// DualAttention for MI455X (gfx1250, wave32, WMMA).
// Heavy path: five 14336x2048x2048 GEMMs on v_wmma_f32_16x16x32_bf16 with
// double-buffered async global->LDS staging (ASYNCcnt pipeline).
// ---------------------------------------------------------------------------

typedef __bf16 bf16_t;
typedef __attribute__((ext_vector_type(16))) __bf16 v16bf;
typedef __attribute__((ext_vector_type(8)))  __bf16 v8bf;
typedef __attribute__((ext_vector_type(8)))  float  v8f;

#define N_VARS 7
#define NPATCH 64
#define PSIZE  16
#define DM     128
#define NH     8
#define PD     2048          // PSIZE*DM
#define BB     224           // BATCH*N_VARS
#define ROWS   14336         // BB*NPATCH
#define OUT_ELEMS 29360128   // BB*NPATCH*PSIZE*DM

// ---- CDNA5 async global->LDS support (compile-safe detection) -------------
#if __has_builtin(__builtin_amdgcn_global_load_async_to_lds_b128)
#define USE_ASYNC_LDS 1
#endif

// Builtin expects: (int4 addrspace(1)*, int4 addrspace(3)*, imm offset, imm cpol)
// (clang prints addrspace(1) as "__device__" in HIP diagnostics).
typedef int v4i_t __attribute__((vector_size(4 * sizeof(int))));
typedef __attribute__((address_space(1))) v4i_t* gptr_t;
typedef __attribute__((address_space(3))) v4i_t* lptr_t;

__device__ static __forceinline__ gptr_t as_global(const void* p) {
  return (gptr_t)(unsigned long long)p;
}
__device__ static __forceinline__ lptr_t as_lds(void* p) {
  // generic LDS pointer: addr[31:0] is the LDS offset (CDNA5 flat addressing)
  return (lptr_t)(unsigned int)(unsigned long long)p;
}

__device__ static __forceinline__ void wait_async_le4() {
#if defined(USE_ASYNC_LDS)
#if __has_builtin(__builtin_amdgcn_s_wait_asynccnt)
  __builtin_amdgcn_s_wait_asynccnt(4);
#else
  asm volatile("s_wait_asynccnt 0x4" ::: "memory");
#endif
#endif
}
__device__ static __forceinline__ void wait_async_le0() {
#if defined(USE_ASYNC_LDS)
#if __has_builtin(__builtin_amdgcn_s_wait_asynccnt)
  __builtin_amdgcn_s_wait_asynccnt(0);
#else
  asm volatile("s_wait_asynccnt 0x0" ::: "memory");
#endif
#endif
}

// ---------------------------------------------------------------------------
// WMMA fragment loaders (wave32).
// A (16x32 bf16, MxK): lane L: m = L&15, half = L>>4; per-lane k indices are
//   {half*8 .. half*8+7} U {16+half*8 .. 16+half*8+7}  -> two 16B loads.
// B (32x16 bf16, KxN): lane L: n = L&15, half = L>>4; per-lane k indices are
//   {half*8 .. half*8+7} U {16+half*8 .. 16+half*8+7}  -> two 16B loads.
// C/D (16x16 f32): lane L: n = L&15; acc[v] is row m = v + 8*(L>>4).
// ---------------------------------------------------------------------------
__device__ static inline v16bf load_frag_a(const bf16_t* base, int stride,
                                           int row0, int k0, int lane) {
  int r = lane & 15, half = lane >> 4;
  const v8bf* p0 = (const v8bf*)(base + (size_t)(row0 + r) * stride + k0 + half * 8);
  const v8bf* p1 = (const v8bf*)(base + (size_t)(row0 + r) * stride + k0 + 16 + half * 8);
  v8bf a0 = *p0, a1 = *p1;
  v16bf f;
#pragma unroll
  for (int i = 0; i < 8; ++i) { f[i] = a0[i]; f[i + 8] = a1[i]; }
  return f;
}

__device__ static inline v16bf load_frag_b(const bf16_t* base, int stride,
                                           int n0, int k0, int lane) {
  int n = lane & 15, half = lane >> 4;
  const v8bf* p0 = (const v8bf*)(base + (size_t)(n0 + n) * stride + k0 + half * 8);
  const v8bf* p1 = (const v8bf*)(base + (size_t)(n0 + n) * stride + k0 + 16 + half * 8);
  v8bf a0 = *p0, a1 = *p1;
  v16bf f;
#pragma unroll
  for (int i = 0; i < 8; ++i) { f[i] = a0[i]; f[i + 8] = a1[i]; }
  return f;
}

__device__ static inline v8f wmma_bf16(v16bf a, v16bf b, v8f c) {
  return __builtin_amdgcn_wmma_f32_16x16x32_bf16(false, a, false, b,
                                                 (short)0, c, false, false);
}

// ---------------------------------------------------------------------------
// sincos positional encoding (single block): raw -> mean/std normalize.
// ---------------------------------------------------------------------------
__global__ __launch_bounds__(256) void sincos_pe_kernel(float* pe, int L, int d) {
  __shared__ float rsum[256], rsq[256];
  int t = threadIdx.x;
  int total = L * d;
  float c = -logf(10000.0f) / (float)d;
  float lsum = 0.f, lsq = 0.f;
  for (int i = t; i < total; i += 256) {
    int row = i / d, col = i % d;
    float div = expf((float)(col & ~1) * c);
    float v = (col & 1) ? cosf((float)row * div) : sinf((float)row * div);
    pe[i] = v; lsum += v; lsq += v * v;
  }
  rsum[t] = lsum; rsq[t] = lsq;
  __syncthreads();
  for (int s = 128; s > 0; s >>= 1) {
    if (t < s) { rsum[t] += rsum[t + s]; rsq[t] += rsq[t + s]; }
    __syncthreads();
  }
  float mean = rsum[0] / (float)total;
  float var  = rsq[0] / (float)total - mean * mean;
  float inv  = 1.0f / (sqrtf(fmaxf(var, 0.f)) * 10.0f);
  for (int i = t; i < total; i += 256) pe[i] = (pe[i] - mean) * inv;
}

// f32 [K][N] -> bf16 transposed [N][K] (LDS tiled transpose)
__global__ __launch_bounds__(256) void transpose_cvt_kernel(const float* __restrict__ W,
                                                            bf16_t* __restrict__ Wt,
                                                            int K, int N) {
  __shared__ bf16_t tile[32][33];
  int t = threadIdx.x, tx = t & 31, ty = t >> 5;
  int k0 = blockIdx.y * 32, n0 = blockIdx.x * 32;
  for (int r = ty; r < 32; r += 8)
    tile[r][tx] = (bf16_t)W[(size_t)(k0 + r) * N + n0 + tx];
  __syncthreads();
  for (int r = ty; r < 32; r += 8)
    Wt[(size_t)(n0 + r) * K + k0 + tx] = tile[tx][r];
}

__global__ __launch_bounds__(256) void cvt_bf16_kernel(const float* __restrict__ src,
                                                       bf16_t* __restrict__ dst,
                                                       int total) {
  int i = blockIdx.x * 256 + threadIdx.x;
  if (i < total) dst[i] = (bf16_t)src[i];
}

// ---------------------------------------------------------------------------
// Tiled bf16 WMMA GEMM: C[M,N] = A[M,K] @ Wt[N,K]^T  (M=14336, K=N=2048)
// Block tile 128x128, K-step 32, 8 waves (4x2 wave grid, 2x4 16x16 tiles each).
// Double-buffered LDS; global->LDS via async-copy (ASYNCcnt) when available.
// MODE 0: +bias +pos_inter[row%64]  -> bf16 out      (inter_embed)
// MODE 1: +bias                     -> bf16 out      (iq/ik/iv)
// MODE 2: +bias +outF (intra)       -> f32  d_out    (inter_out, fused add)
// ---------------------------------------------------------------------------
#define LDA 40  // padded LDS stride (bf16 elems): 80B, 16B-aligned

__device__ static __forceinline__ void stage_tile_async(
    const bf16_t* ga, const bf16_t* gb, bf16_t* la, bf16_t* lb) {
#if defined(USE_ASYNC_LDS)
  __builtin_amdgcn_global_load_async_to_lds_b128(as_global(ga),     as_lds(la),     0, 0);
  __builtin_amdgcn_global_load_async_to_lds_b128(as_global(ga + 8), as_lds(la + 8), 0, 0);
  __builtin_amdgcn_global_load_async_to_lds_b128(as_global(gb),     as_lds(lb),     0, 0);
  __builtin_amdgcn_global_load_async_to_lds_b128(as_global(gb + 8), as_lds(lb + 8), 0, 0);
#else
  const v8bf* pa = (const v8bf*)ga; const v8bf* pb = (const v8bf*)gb;
  v8bf a0 = pa[0], a1 = pa[1], b0 = pb[0], b1 = pb[1];
  ((v8bf*)la)[0] = a0; ((v8bf*)la)[1] = a1;
  ((v8bf*)lb)[0] = b0; ((v8bf*)lb)[1] = b1;
#endif
}

__device__ static __forceinline__ void compute_tile(
    const bf16_t* As, const bf16_t* Bs, int wm, int wn, int lane,
    v8f (&acc)[2][4]) {
  v16bf af[2], bfb[4];
#pragma unroll
  for (int i = 0; i < 2; ++i) af[i] = load_frag_a(As, LDA, wm + i * 16, 0, lane);
#pragma unroll
  for (int j = 0; j < 4; ++j) bfb[j] = load_frag_b(Bs, LDA, wn + j * 16, 0, lane);
#pragma unroll
  for (int i = 0; i < 2; ++i)
#pragma unroll
    for (int j = 0; j < 4; ++j) acc[i][j] = wmma_bf16(af[i], bfb[j], acc[i][j]);
}

template <int MODE>
__global__ __launch_bounds__(256) void gemm_bf16_kernel(
    const bf16_t* __restrict__ A, const bf16_t* __restrict__ Bt,
    const float* __restrict__ bias, const float* __restrict__ pos,
    float* __restrict__ outF, bf16_t* __restrict__ outB,
    int K, int N) {
  __shared__ bf16_t As[2][128 * LDA];
  __shared__ bf16_t Bs[2][128 * LDA];

  int t = threadIdx.x;
  int wid = t >> 5, lane = t & 31;
  int bm = blockIdx.y * 128, bn = blockIdx.x * 128;
  int wm = (wid >> 1) * 32, wn = (wid & 1) * 64;

  v8f zero = {0.f, 0.f, 0.f, 0.f, 0.f, 0.f, 0.f, 0.f};
  v8f acc[2][4];
#pragma unroll
  for (int i = 0; i < 2; ++i)
#pragma unroll
    for (int j = 0; j < 4; ++j) acc[i][j] = zero;

  int lrow = t >> 1;            // 0..127
  int lch  = (t & 1) * 16;      // chunk of 16 bf16 (32B)
  const bf16_t* gaBase = A  + (size_t)(bm + lrow) * K + lch;
  const bf16_t* gbBase = Bt + (size_t)(bn + lrow) * K + lch;
  bf16_t* la[2] = {As[0] + lrow * LDA + lch, As[1] + lrow * LDA + lch};
  bf16_t* lb[2] = {Bs[0] + lrow * LDA + lch, Bs[1] + lrow * LDA + lch};

  int nT = K / 32;
  // Pipelined: stage tile i+1 while computing tile i.
  stage_tile_async(gaBase, gbBase, la[0], lb[0]);
  for (int i = 0; i < nT - 1; ++i) {
    int cur = i & 1;
    stage_tile_async(gaBase + (i + 1) * 32, gbBase + (i + 1) * 32,
                     la[cur ^ 1], lb[cur ^ 1]);
    wait_async_le4();           // tile i landed (async loads complete in order)
    __syncthreads();
    compute_tile(As[cur], Bs[cur], wm, wn, lane, acc);
    __syncthreads();            // all waves done reading before buf reuse
  }
  {
    int cur = (nT - 1) & 1;
    wait_async_le0();
    __syncthreads();
    compute_tile(As[cur], Bs[cur], wm, wn, lane, acc);
  }

  int r = lane & 15, half = lane >> 4;
#pragma unroll
  for (int i = 0; i < 2; ++i)
#pragma unroll
    for (int j = 0; j < 4; ++j)
#pragma unroll
      for (int v = 0; v < 8; ++v) {
        int gm = bm + wm + i * 16 + v + 8 * half;
        int gn = bn + wn + j * 16 + r;
        size_t idx = (size_t)gm * N + gn;
        float c = acc[i][j][v] + bias[gn];
        if (MODE == 0) {
          c += pos[(size_t)(gm & 63) * N + gn];
          outB[idx] = (bf16_t)c;
        } else if (MODE == 1) {
          outB[idx] = (bf16_t)c;
        } else {
          outF[idx] = c + outF[idx];   // add intra term already in d_out
        }
      }
}

// ---------------------------------------------------------------------------
// Inter attention: one block per (b,h). scores = Q K^T * 0.25, softmax (-> attn2
// output), O = attn @ V. Q/K fragments straight from global; V transposed to LDS.
// ---------------------------------------------------------------------------
__global__ __launch_bounds__(256) void inter_attn_kernel(
    const bf16_t* __restrict__ q2, const bf16_t* __restrict__ k2,
    const bf16_t* __restrict__ v2, bf16_t* __restrict__ o2,
    float* __restrict__ attn_out) {
  __shared__ float  sc[64 * 64];      // 16 KB
  __shared__ bf16_t attnb[64 * 72];   // 9 KB  (padded stride 72)
  __shared__ bf16_t Vt[256 * 72];     // 36 KB (Vt[d][k])

  int h = blockIdx.x, b = blockIdx.y;
  int t = threadIdx.x, wid = t >> 5, lane = t & 31;
  int r = lane & 15, half = lane >> 4;

  const bf16_t* Q  = q2 + (size_t)b * 64 * PD + h * 256;  // row stride PD
  const bf16_t* Km = k2 + (size_t)b * 64 * PD + h * 256;
  const bf16_t* V  = v2 + (size_t)b * 64 * PD + h * 256;

  // Phase 1: scores (64x64). 16 tiles / 8 waves = 2 per wave.
  int ti = wid >> 1;
  int tj0 = (wid & 1) * 2;
  v8f zero = {0.f, 0.f, 0.f, 0.f, 0.f, 0.f, 0.f, 0.f};
  v8f acc1[2] = {zero, zero};
  for (int kk = 0; kk < 256; kk += 32) {
    v16bf aq = load_frag_a(Q, PD, ti * 16, kk, lane);
#pragma unroll
    for (int jj = 0; jj < 2; ++jj) {
      v16bf bk = load_frag_b(Km, PD, (tj0 + jj) * 16, kk, lane);  // K^T
      acc1[jj] = wmma_bf16(aq, bk, acc1[jj]);
    }
  }
#pragma unroll
  for (int jj = 0; jj < 2; ++jj)
#pragma unroll
    for (int v = 0; v < 8; ++v)
      sc[(ti * 16 + v + 8 * half) * 64 + (tj0 + jj) * 16 + r] = acc1[jj][v] * 0.25f;

  // Transpose V into LDS: thread t owns column d=t (coalesced global reads).
  {
    const bf16_t* vp = V + t;
    for (int k = 0; k < 64; ++k) Vt[t * 72 + k] = vp[(size_t)k * PD];
  }
  __syncthreads();

  // Softmax rows; write attn2 (f32 output) and bf16 copy for the O GEMM.
  if (t < 64) {
    float m = sc[t * 64];
    for (int j = 1; j < 64; ++j) m = fmaxf(m, sc[t * 64 + j]);
    float s = 0.f;
    for (int j = 0; j < 64; ++j) { float e = expf(sc[t * 64 + j] - m); sc[t * 64 + j] = e; s += e; }
    float inv = 1.0f / s;
    float* ao = attn_out + (((size_t)b * NH + h) * 64 + t) * 64;
    for (int j = 0; j < 64; ++j) {
      float a = sc[t * 64 + j] * inv;
      ao[j] = a;
      attnb[t * 72 + j] = (bf16_t)a;
    }
  }
  __syncthreads();

  // Phase 2: O = attn(64x64) @ V(64x256). 64 tiles / 8 waves = 8 per wave.
  int oi = wid >> 1;
  int jb = (wid & 1) * 8;
  v8f acc2[8];
#pragma unroll
  for (int j = 0; j < 8; ++j) acc2[j] = zero;
  for (int kk = 0; kk < 64; kk += 32) {
    v16bf aa = load_frag_a(attnb, 72, oi * 16, kk, lane);
#pragma unroll
    for (int j = 0; j < 8; ++j) {
      v16bf bb = load_frag_b(Vt, 72, (jb + j) * 16, kk, lane);
      acc2[j] = wmma_bf16(aa, bb, acc2[j]);
    }
  }
#pragma unroll
  for (int j = 0; j < 8; ++j)
#pragma unroll
    for (int v = 0; v < 8; ++v) {
      int m = oi * 16 + v + 8 * half;
      int n = (jb + j) * 16 + r;
      o2[((size_t)b * 64 + m) * PD + h * 256 + n] = (bf16_t)acc2[j][v];
    }
}

// ---------------------------------------------------------------------------
// Intra path, fused: token embed + q/k/v proj + 8-head attention over 16 keys.
// One block (128 threads) per (b,n). Writes z[b, d, n].
// ---------------------------------------------------------------------------
__global__ __launch_bounds__(128) void intra_fused_kernel(
    const float* __restrict__ x, const float* __restrict__ tokens,
    const float* __restrict__ eW, const float* __restrict__ eb,
    const float* __restrict__ qW, const float* __restrict__ qb,
    const float* __restrict__ kW, const float* __restrict__ kb,
    const float* __restrict__ vW, const float* __restrict__ vb,
    const float* __restrict__ posq, const float* __restrict__ poskv,
    float* __restrict__ zbuf) {
  __shared__ float tq[128], qv[128], scs[128], att[128];
  __shared__ float cx[16 * 128], kk_[16 * 128], vv[16 * 128];

  int n = blockIdx.x, b = blockIdx.y, t = threadIdx.x;
  int c = b % N_VARS;

  // token embed (+pos_q)
  const float* tok = tokens + ((size_t)c * NPATCH + n) * 16;
  float s = 0.f;
  for (int f = 0; f < 16; ++f) s += tok[f] * eW[((size_t)n * 16 + f) * 128 + t];
  tq[t] = s + eb[(size_t)n * 128 + t] + posq[t];
  __syncthreads();

  // q projection
  float q = qb[t];
  for (int k = 0; k < 128; ++k) q += tq[k] * qW[(size_t)k * 128 + t];
  qv[t] = q;

  // cx = x + pos_kv
  const float* xb = x + ((size_t)b * NPATCH + n) * PSIZE * 128;
  for (int p = 0; p < 16; ++p) cx[p * 128 + t] = xb[p * 128 + t] + poskv[p * 128 + t];
  __syncthreads();

  // k,v projections
  for (int p = 0; p < 16; ++p) {
    float ak = kb[t], av = vb[t];
    for (int k = 0; k < 128; ++k) {
      float cvv = cx[p * 128 + k];
      ak += cvv * kW[(size_t)k * 128 + t];
      av += cvv * vW[(size_t)k * 128 + t];
    }
    kk_[p * 128 + t] = ak;
    vv[p * 128 + t]  = av;
  }
  __syncthreads();

  // scores (scale = 1/sqrt(D//H) = 0.25)
  int hh = t >> 4, pp = t & 15;
  float sc2 = 0.f;
  for (int d = 0; d < 16; ++d) sc2 += qv[hh * 16 + d] * kk_[pp * 128 + hh * 16 + d];
  scs[t] = sc2 * 0.25f;
  __syncthreads();

  if (t < 8) {
    float m = scs[t * 16];
    for (int j = 1; j < 16; ++j) m = fmaxf(m, scs[t * 16 + j]);
    float sum = 0.f;
    for (int j = 0; j < 16; ++j) { float e = expf(scs[t * 16 + j] - m); att[t * 16 + j] = e; sum += e; }
    float inv = 1.0f / sum;
    for (int j = 0; j < 16; ++j) att[t * 16 + j] *= inv;
  }
  __syncthreads();

  float o = 0.f;
  for (int p = 0; p < 16; ++p) o += att[hh * 16 + p] * vv[p * 128 + t];
  zbuf[((size_t)b * 128 + t) * NPATCH + n] = o;  // z[b, d=t, n]
}

// intra out: out[b,np,p,d] = sum_k z[b,d,k] * W[k, np*16+p] + bias  (writes d_out)
__global__ __launch_bounds__(256) void intra_out_kernel(
    const float* __restrict__ zbuf, const float* __restrict__ W,
    const float* __restrict__ bias, float* __restrict__ out) {
  __shared__ float zl[128 * 64];  // z[b, :, :] 32KB
  int np = blockIdx.x, b = blockIdx.y, t = threadIdx.x;
  for (int i = t; i < 128 * 64; i += 256) zl[i] = zbuf[(size_t)b * 128 * 64 + i];
  __syncthreads();
  int dm = t & 127, ph = t >> 7;  // ph in {0,1}
  for (int pi = 0; pi < 8; ++pi) {
    int p = ph * 8 + pi;
    int col = np * 16 + p;
    float a = bias[col];
    for (int k = 0; k < 64; ++k) a += zl[dm * 64 + k] * W[(size_t)k * 1024 + col];
    out[(((size_t)b * NPATCH + np) * PSIZE + p) * 128 + dm] = a;
  }
}

// ---------------------------------------------------------------------------
extern "C" void kernel_launch(void* const* d_in, const int* in_sizes, int n_in,
                              void* d_out, int out_size, void* d_ws, size_t ws_size,
                              hipStream_t stream) {
  const float* x        = (const float*)d_in[0];
  const float* tokens   = (const float*)d_in[1];
  const float* embed_W  = (const float*)d_in[2];
  const float* embed_b  = (const float*)d_in[3];
  const float* q_W = (const float*)d_in[4];  const float* q_b = (const float*)d_in[5];
  const float* k_W = (const float*)d_in[6];  const float* k_b = (const float*)d_in[7];
  const float* v_W = (const float*)d_in[8];  const float* v_b = (const float*)d_in[9];
  const float* intra_out_W = (const float*)d_in[10];
  const float* intra_out_b = (const float*)d_in[11];
  const float* ie_W = (const float*)d_in[12]; const float* ie_b = (const float*)d_in[13];
  const float* iq_W = (const float*)d_in[14]; const float* iq_b = (const float*)d_in[15];
  const float* ik_W = (const float*)d_in[16]; const float* ik_b = (const float*)d_in[17];
  const float* iv_W = (const float*)d_in[18]; const float* iv_b = (const float*)d_in[19];
  const float* io_W = (const float*)d_in[20]; const float* io_b = (const float*)d_in[21];

  float* out   = (float*)d_out;
  float* attn2 = out + OUT_ELEMS;

  // ---- workspace carve (all 256B aligned) ----
  char* w = (char*)d_ws;
  size_t off = 0;
  auto carve = [&](size_t bytes) { void* p = w + off; off += (bytes + 255) & ~(size_t)255; return p; };
  float*  posq     = (float*)carve(128 * 4);
  float*  poskv    = (float*)carve(PSIZE * DM * 4);
  float*  posinter = (float*)carve((size_t)NPATCH * PD * 4);
  float*  zbuf     = (float*)carve((size_t)BB * 128 * 64 * 4);
  bf16_t* WtE = (bf16_t*)carve((size_t)PD * PD * 2);
  bf16_t* WtQ = (bf16_t*)carve((size_t)PD * PD * 2);
  bf16_t* WtK = (bf16_t*)carve((size_t)PD * PD * 2);
  bf16_t* WtV = (bf16_t*)carve((size_t)PD * PD * 2);
  bf16_t* WtO = (bf16_t*)carve((size_t)PD * PD * 2);
  bf16_t* bufA = (bf16_t*)carve((size_t)ROWS * PD * 2);  // x_bf -> q2
  bf16_t* bufB = (bf16_t*)carve((size_t)ROWS * PD * 2);  // xr_bf -> o2
  bf16_t* bufC = (bf16_t*)carve((size_t)ROWS * PD * 2);  // k2
  bf16_t* bufD = (bf16_t*)carve((size_t)ROWS * PD * 2);  // v2
  (void)ws_size;

  // ---- positional encodings ----
  sincos_pe_kernel<<<1, 256, 0, stream>>>(posq, 1, DM);
  sincos_pe_kernel<<<1, 256, 0, stream>>>(poskv, PSIZE, DM);
  sincos_pe_kernel<<<1, 256, 0, stream>>>(posinter, NPATCH, PD);

  // ---- weight transpose+convert to bf16, x convert ----
  dim3 tg(PD / 32, PD / 32);
  transpose_cvt_kernel<<<tg, 256, 0, stream>>>(ie_W, WtE, PD, PD);
  transpose_cvt_kernel<<<tg, 256, 0, stream>>>(iq_W, WtQ, PD, PD);
  transpose_cvt_kernel<<<tg, 256, 0, stream>>>(ik_W, WtK, PD, PD);
  transpose_cvt_kernel<<<tg, 256, 0, stream>>>(iv_W, WtV, PD, PD);
  transpose_cvt_kernel<<<tg, 256, 0, stream>>>(io_W, WtO, PD, PD);
  cvt_bf16_kernel<<<OUT_ELEMS / 256, 256, 0, stream>>>(x, bufA, OUT_ELEMS);

  // ---- intra path (writes d_out out-region fully) ----
  intra_fused_kernel<<<dim3(NPATCH, BB), 128, 0, stream>>>(
      x, tokens, embed_W, embed_b, q_W, q_b, k_W, k_b, v_W, v_b,
      posq, poskv, zbuf);
  intra_out_kernel<<<dim3(NPATCH, BB), 256, 0, stream>>>(zbuf, intra_out_W, intra_out_b, out);

  // ---- inter path: five big WMMA GEMMs + attention ----
  dim3 gg(PD / 128, ROWS / 128);  // (16, 112)
  // xr = x @ We + be + pos_inter  -> bf16
  gemm_bf16_kernel<0><<<gg, 256, 0, stream>>>(bufA, WtE, ie_b, posinter, nullptr, bufB, PD, PD);
  // q2/k2/v2 = xr @ W + b -> bf16  (q2 reuses bufA: x_bf is dead after embed)
  gemm_bf16_kernel<1><<<gg, 256, 0, stream>>>(bufB, WtQ, iq_b, nullptr, nullptr, bufA, PD, PD);
  gemm_bf16_kernel<1><<<gg, 256, 0, stream>>>(bufB, WtK, ik_b, nullptr, nullptr, bufC, PD, PD);
  gemm_bf16_kernel<1><<<gg, 256, 0, stream>>>(bufB, WtV, iv_b, nullptr, nullptr, bufD, PD, PD);
  // attention: attn2 -> d_out tail, o2 -> bufB (xr is dead)
  inter_attn_kernel<<<dim3(NH, BB), 256, 0, stream>>>(bufA, bufC, bufD, bufB, attn2);
  // out = intra(d_out) + o2 @ Wo + bo   (f32, in place)
  gemm_bf16_kernel<2><<<gg, 256, 0, stream>>>(bufB, WtO, io_b, nullptr, out, nullptr, PD, PD);
}
